// EdgeDecoder_15831249453675
// MI455X (gfx1250) — matrix operand order repaired and checked by reading, not compiled, not run on hardware
//
#include <hip/hip_runtime.h>
#include <hip/hip_fp16.h>
#include <math.h>

typedef __attribute__((ext_vector_type(16))) _Float16 v16h;
typedef __attribute__((ext_vector_type(8)))  float    v8f;
typedef __attribute__((ext_vector_type(4)))  float    f32x4;

#define HID      128
#define NT_TILES 8            // 128 / 16
#define KT_TILES 8            // 256 / 32

// LDS holds W1 as f16 WMMA-B fragments: [kt][nt][lane][16 halves] = 64 KB
#define B_SLOTS (KT_TILES * NT_TILES * 32)

__global__ __launch_bounds__(256) void edge_decoder_wmma(
    const float* __restrict__ z,    // [N_NODES][128] f32
    const int*   __restrict__ eli,  // [2][E] int32
    const float* __restrict__ w1,   // [256][128] f32 (row-major, K x N)
    const float* __restrict__ b1,   // [128]
    const float* __restrict__ w2,   // [128]
    const float* __restrict__ b2,   // [1]
    float* __restrict__ out,        // [E]
    int E)
{
    __shared__ alignas(32) _Float16 Bs[B_SLOTS * 16];

    const int tid  = threadIdx.x;
    const int lane = tid & 31;
    const int wave = tid >> 5;           // 0..7 (8 wave32 per block)

    // ---- Stage W1 f32 -> f16 into WMMA-B fragment layout in LDS ----
    // Dense 16-bit B (32x16): lane L holds column N = L&15;
// lanes 0-15 carry K = kbase+0..7,16..23, lanes 16-31 carry K = kbase+8..15,24..31.
    for (int s = tid; s < B_SLOTS; s += 256) {
        const int tile = s >> 5;                 // kt*8 + nt
        const int ls   = s & 31;
        const int kt   = tile >> 3;
        const int nt   = tile & 7;
        const int n    = nt * 16 + (ls & 15);
        const int kb   = kt * 32 + ((ls >> 4) << 3);   // +0 or +8
        _Float16* dst  = &Bs[s * 16];
        #pragma unroll
        for (int j = 0; j < 16; ++j)
            dst[j] = (_Float16)w1[(kb + (j & 7) + ((j >> 3) << 4)) * HID + n];
    }
    __syncthreads();

    // Per-lane output-layer constants: C layout has N = lane&15 in every VGPR.
    const int nIdx = lane & 15;
    float b1v[NT_TILES], w2v[NT_TILES];
    #pragma unroll
    for (int nt = 0; nt < NT_TILES; ++nt) {
        b1v[nt] = b1[nt * 16 + nIdx];
        w2v[nt] = w2[nt * 16 + nIdx];
    }
    const float bias2 = b2[0];

    const v16h* Bfrag = reinterpret_cast<const v16h*>(Bs);

    const int tilesTotal = (E + 127) >> 7;       // 128 edges per block-tile
    for (int bt = blockIdx.x; bt < tilesTotal; bt += (int)gridDim.x) {
        const int edgeBase = bt * 128 + wave * 16;     // 16 edges per wave
        const int m  = lane & 15;                      // A row (M)
        int e = edgeBase + m;
        if (e >= E) e = E - 1;                         // clamp gather (tail safety)
        const long long srcRow = (long long)eli[e] * HID;
        const long long dstRow = (long long)eli[E + e] * HID;
        const int hi8 = (lane >> 4) << 3;              // lanes>=16 start at K+8

        // 8 independent accumulator chains (one per 16-col N tile): 64 VGPRs.
        v8f cAll[NT_TILES];
        #pragma unroll
        for (int nt = 0; nt < NT_TILES; ++nt) cAll[nt] = (v8f){};

        // ---- Runtime K loop: gather one A fragment, feed 8 WMMA chains ----
        // Keeping kt as a real loop bounds live registers to
        // 64 (C) + 8 (A) + ~64 (B hoisted) + misc: no spills.
        #pragma unroll 1
        for (int kt = 0; kt < KT_TILES; ++kt) {
            // A fragment 16x32 (f32 -> f16):
            // lane halves 0..7 = K f0..f0+7, halves 8..15 = K f0+16..f0+23,
            // f0 = kt*32 + hi8; K<128 -> src embedding, else dst embedding.
            const bool lo = (kt < 4);
            const long long rb = lo ? srcRow : dstRow;
            const int f0 = kt * 32 + hi8 - (lo ? 0 : HID);
            const float* p0 = z + rb + f0;
            f32x4 c0 = *(const f32x4*)(p0);
            f32x4 c1 = *(const f32x4*)(p0 + 4);
            f32x4 c2 = *(const f32x4*)(p0 + 16);
            f32x4 c3 = *(const f32x4*)(p0 + 20);
            v16h a;
            #pragma unroll
            for (int j = 0; j < 4; ++j) {
                a[j]      = (_Float16)c0[j];
                a[4 + j]  = (_Float16)c1[j];
                a[8 + j]  = (_Float16)c2[j];
                a[12 + j] = (_Float16)c3[j];
            }
            #pragma unroll
            for (int nt = 0; nt < NT_TILES; ++nt) {
                v16h b = Bfrag[(kt * NT_TILES + nt) * 32 + lane];
                cAll[nt] = __builtin_amdgcn_wmma_f32_16x16x32_f16(
                        /*neg_a=*/false, a, /*neg_b=*/false, b,
                        /*c_mod=*/(short)0, cAll[nt],
                        /*reuse_a=*/false, /*reuse_b=*/false);
            }
        }

        // ---- Fused layer 2: +b1, relu, * w2, partial dot over N ----
        float acc[8];
        #pragma unroll
        for (int j = 0; j < 8; ++j) acc[j] = 0.0f;

        #pragma unroll
        for (int nt = 0; nt < NT_TILES; ++nt) {
            const float bias = b1v[nt];
            const float wv   = w2v[nt];
            #pragma unroll
            for (int j = 0; j < 8; ++j) {
                float h = cAll[nt][j] + bias;
                h = h > 0.0f ? h : 0.0f;
                acc[j] = fmaf(h, wv, acc[j]);
            }
        }

        // ---- Reduce over the 16 N-lanes of each half-wave ----
        // (xor masks 1,2,4,8 stay within lanes 0-15 / 16-31; those groups hold
        //  rows M = j and M = 8+j respectively, matching the C layout.)
        #pragma unroll
        for (int j = 0; j < 8; ++j) {
            float v = acc[j];
            v += __shfl_xor(v, 1, 32);
            v += __shfl_xor(v, 2, 32);
            v += __shfl_xor(v, 4, 32);
            v += __shfl_xor(v, 8, 32);
            acc[j] = v;
        }

        if ((lane & 15) == 0) {
            const int rbase = edgeBase + ((lane >> 4) << 3);  // rows 0-7 / 8-15
            #pragma unroll
            for (int j = 0; j < 8; ++j) {
                const int eo = rbase + j;
                if (eo < E) {
                    const float v = acc[j] + bias2;
                    out[eo] = 1.0f / (1.0f + __expf(-v));     // sigmoid
                }
            }
        }
    }
}

extern "C" void kernel_launch(void* const* d_in, const int* in_sizes, int n_in,
                              void* d_out, int out_size, void* d_ws, size_t ws_size,
                              hipStream_t stream) {
    const float* z   = (const float*)d_in[0];
    const int*   eli = (const int*)d_in[1];   // [2, E] indices
    const float* w1  = (const float*)d_in[2];
    const float* b1  = (const float*)d_in[3];
    const float* w2  = (const float*)d_in[4];
    const float* b2  = (const float*)d_in[5];
    float* out = (float*)d_out;

    const int E = in_sizes[1] / 2;
    const int tiles = (E + 127) >> 7;          // 128 edges per block
    int grid = tiles < 1024 ? tiles : 1024;    // grid-stride amortizes W1 staging
    if (grid < 1) grid = 1;

    edge_decoder_wmma<<<grid, 256, 0, stream>>>(z, eli, w1, b1, w2, b2, out, E);
}
